// KPConvSimpleBlock_25056839205743
// MI455X (gfx1250) — matrix operand order repaired and checked by reading, not compiled, not run on hardware
//
#include <hip/hip_runtime.h>

// ---------------------------------------------------------------------------
// KPConv (15 kernel points) + FastBatchNorm1d (batch stats) + LeakyReLU(0.2)
// fused for MI455X (gfx1250, wave32, WMMA bf16 16x16x32, f32 accumulate).
// ---------------------------------------------------------------------------

typedef __bf16 v16bf __attribute__((ext_vector_type(16)));
typedef __bf16 v8bf  __attribute__((ext_vector_type(8)));
typedef __bf16 v2bf  __attribute__((ext_vector_type(2)));
typedef float  v8f   __attribute__((ext_vector_type(8)));

#define NPTS   100000
#define NNB    32
#define CIN    64
#define COUT   128
#define KP     15
#define KRP    1024          // padded flattened contraction dim (permuted: r' = c*16 + k)
#define KSTEPS (KRP / 32)    // 32 WMMA k-steps of 32
#define BN_EPS 1e-5f
#define NEG_SLOPE 0.2f

__device__ __forceinline__ __bf16 tobf(float x) { return (__bf16)x; }

// packed f32x2 -> bf16x2 (hardware v_cvt_pk_bf16_f32 when available)
__device__ __forceinline__ unsigned int pkbf(float a, float b) {
#if __has_builtin(__builtin_amdgcn_cvt_pk_bf16_f32)
  return __builtin_bit_cast(unsigned int, __builtin_amdgcn_cvt_pk_bf16_f32(a, b));
#else
  v2bf p; p[0] = (__bf16)a; p[1] = (__bf16)b;
  return __builtin_bit_cast(unsigned int, p);
#endif
}
__device__ __forceinline__ v16bf bfcat(v8bf lo, v8bf hi) {
  return __builtin_shufflevector(lo, hi, 0,1,2,3,4,5,6,7,8,9,10,11,12,13,14,15);
}
// per-wave LDS producer->consumer: DS ops are in-order within a wave; this
// stops the compiler from reordering across the phase boundary.
#define LDS_BAR() asm volatile("s_wait_dscnt 0" ::: "memory")

// ---------------------------------------------------------------------------
// Kernel 0: weight (960x128 f32, rows r = k*64+c) -> bf16 rows r' = c*16+k,
// zero-padded to 1024x128; zero BN accumulators.
// ---------------------------------------------------------------------------
__global__ __launch_bounds__(256) void kpconv_prep(const float* __restrict__ w,
                                                   __bf16* __restrict__ wbf,
                                                   float* __restrict__ stats) {
  int i = blockIdx.x * blockDim.x + threadIdx.x;
  if (i < 2 * COUT) stats[i] = 0.0f;
  if (i < KRP * COUT) {
    int rp = i >> 7;        // permuted row  r' = c*16 + k
    int o  = i & (COUT - 1);
    int k  = rp & 15;
    int c  = rp >> 4;
    wbf[i] = (k < KP) ? tobf(w[(k * CIN + c) * COUT + o]) : tobf(0.0f);
  }
}

// ---------------------------------------------------------------------------
// Kernel 1: per block = 16 points (4 waves).
//   Stage 1 (per wave, 4 points): influence (16x32 bf16) x neigh feats
//     (32x64 bf16) via 4x WMMA -> agg (16x64) -> bf16 LDS rows=points,
//     contraction index permuted so each D fragment stores as one b128.
//   Stage 2: (16 x 1024) @ (1024 x 128) via 32 ksteps x 2 ntiles per wave.
//   Epilogue: raw out + per-channel sum/sumsq atomics for BN.
// ---------------------------------------------------------------------------
__global__ __launch_bounds__(128) void kpconv_main(
    const float* __restrict__ feats,
    const float* __restrict__ xyz,
    const int*   __restrict__ nbr,
    const float* __restrict__ kpts,
    const __bf16* __restrict__ wbf,
    float* __restrict__ out,
    float* __restrict__ bnsum,
    float* __restrict__ bnsq)
{
  __shared__ alignas(32) __bf16 aggS[16 * KRP];     // 32 KB: agg bf16, rows = points
  __shared__ alignas(32) __bf16 a4S[4][16 * NNB];   //  4 KB: influence per wave
  __shared__ alignas(32) __bf16 b4S[4][NNB * CIN];  // 16 KB: neighbor feats per wave

  const int lane = threadIdx.x & 31;
  const int wave = threadIdx.x >> 5;
  const int r    = lane & 15;   // row within A / N within B halves
  const int half = lane >> 4;   // 0: lanes 0-15, 1: lanes 16-31
  const int n0   = blockIdx.x * 16;

  float kx[KP], ky[KP], kz[KP];
#pragma unroll
  for (int k = 0; k < KP; ++k) {
    kx[k] = kpts[3 * k + 0];
    ky[k] = kpts[3 * k + 1];
    kz[k] = kpts[3 * k + 2];
  }

  __bf16* A4 = a4S[wave];
  __bf16* B4 = b4S[wave];

  // ---------------- Stage 1: aggregation per point ----------------
  for (int pp = 0; pp < 4; ++pp) {
    const int p = wave * 4 + pp;
    const int n = n0 + p;
    LDS_BAR();  // WAR: previous iteration's WMMA reads vs this iteration's stores

    // lane = neighbor index e
    int idx = nbr[n * NNB + lane];
    const bool valid = ((unsigned)idx < (unsigned)NPTS);  // shadow-point guard
    const int ic = valid ? idx : 0;

    const float cx = xyz[3 * n + 0], cy = xyz[3 * n + 1], cz = xyz[3 * n + 2];
    const float rx = xyz[3 * ic + 0] - cx;
    const float ry = xyz[3 * ic + 1] - cy;
    const float rz = xyz[3 * ic + 2] - cz;

    // influence matrix A4[k][e], row 15 zero pad
#pragma unroll
    for (int k = 0; k < KP; ++k) {
      float dx = rx - kx[k], dy = ry - ky[k], dz = rz - kz[k];
      float dist = sqrtf(dx * dx + dy * dy + dz * dz);
      float infl = fmaxf(1.0f - dist * 0.5f, 0.0f);   // POINT_INFLUENCE = 2.0
      if (!valid) infl = 0.0f;
      A4[k * NNB + lane] = tobf(infl);
    }
    A4[KP * NNB + lane] = tobf(0.0f);

    // gather neighbor features (row e = lane), f32 -> bf16, 16B vector stores
    const float* frow = feats + (size_t)ic * CIN;
#pragma unroll
    for (int c = 0; c < CIN; c += 8) {
      float4 va = *(const float4*)(frow + c);
      float4 vb = *(const float4*)(frow + c + 4);
      uint4 s;
      s.x = pkbf(va.x, va.y);
      s.y = pkbf(va.z, va.w);
      s.z = pkbf(vb.x, vb.y);
      s.w = pkbf(vb.z, vb.w);
      *(uint4*)(B4 + lane * CIN + c) = s;
    }
    LDS_BAR();  // RAW: stores above -> WMMA fragment loads below

    // A fragment: lanes 0-15 row r: K 0-7 & 16-23; lanes 16-31: K 8-15 & 24-31
    v8bf alo = *(const v8bf*)(A4 + r * NNB + half * 8);
    v8bf ahi = *(const v8bf*)(A4 + r * NNB + 16 + half * 8);
    v16bf afrag = bfcat(alo, ahi);
    const int kb = half * 8;
#pragma unroll
    for (int t = 0; t < 4; ++t) {
      // B fragment: lane = N column (t*16 + r), elements = K rows in table order
      v16bf bfrag;
#pragma unroll
      for (int s = 0; s < 16; ++s) {
        const int e = (s < 8) ? (kb + s) : (16 + kb + (s - 8));
        bfrag[s] = B4[e * CIN + t * 16 + r];
      }
      v8f acc = {0.f, 0.f, 0.f, 0.f, 0.f, 0.f, 0.f, 0.f};
      acc = __builtin_amdgcn_wmma_f32_16x16x32_bf16(
          false, afrag, false, bfrag, (short)0, acc, false, false);
      // D elem j, lane L: k = j + (L<16?0:8), c = t*16 + (L%16)
      // permuted flatten r' = c*16 + k  ->  8 adjacent bf16 = one 16B store
      const int c = t * 16 + r;
      uint4 s;
      s.x = pkbf(acc[0], acc[1]);
      s.y = pkbf(acc[2], acc[3]);
      s.z = pkbf(acc[4], acc[5]);
      s.w = pkbf(acc[6], acc[7]);
      *(uint4*)(aggS + p * KRP + c * 16 + kb) = s;
    }
  }

  __syncthreads();

  // ---------------- Stage 2: (16 x 1024) @ (1024 x 128) ----------------
  const int nt0 = wave * 2, nt1 = nt0 + 1;  // two 16-col output tiles per wave
  v8f acc0 = {0.f, 0.f, 0.f, 0.f, 0.f, 0.f, 0.f, 0.f};
  v8f acc1 = {0.f, 0.f, 0.f, 0.f, 0.f, 0.f, 0.f, 0.f};
#pragma unroll 4
  for (int ks = 0; ks < KSTEPS; ++ks) {
    const __bf16* ap = aggS + r * KRP + ks * 32 + half * 8;
    v16bf afrag = bfcat(*(const v8bf*)ap, *(const v8bf*)(ap + 16));
    v16bf b0, b1;
#pragma unroll
    for (int s = 0; s < 16; ++s) {
      const int kk = ks * 32 + ((s < 8) ? (half * 8 + s) : (16 + half * 8 + (s - 8)));
      b0[s] = wbf[(size_t)kk * COUT + nt0 * 16 + r];
      b1[s] = wbf[(size_t)kk * COUT + nt1 * 16 + r];
    }
    acc0 = __builtin_amdgcn_wmma_f32_16x16x32_bf16(
        false, afrag, false, b0, (short)0, acc0, false, false);
    acc1 = __builtin_amdgcn_wmma_f32_16x16x32_bf16(
        false, afrag, false, b1, (short)0, acc1, false, false);
  }

  // ---------------- Epilogue: raw out + BN partial sums ----------------
  const int mb = half * 8;
  float s0 = 0.f, q0 = 0.f, s1 = 0.f, q1 = 0.f;
#pragma unroll
  for (int j = 0; j < 8; ++j) {
    float v0 = acc0[j], v1 = acc1[j];
    out[(size_t)(n0 + mb + j) * COUT + nt0 * 16 + r] = v0;
    out[(size_t)(n0 + mb + j) * COUT + nt1 * 16 + r] = v1;
    s0 += v0; q0 += v0 * v0;
    s1 += v1; q1 += v1 * v1;
  }
  // lanes L and L^16 hold the same channel, complementary point halves
  s0 += __shfl_xor(s0, 16); q0 += __shfl_xor(q0, 16);
  s1 += __shfl_xor(s1, 16); q1 += __shfl_xor(q1, 16);
  if (half == 0) {
    atomicAdd(&bnsum[nt0 * 16 + r], s0);
    atomicAdd(&bnsq [nt0 * 16 + r], q0);
    atomicAdd(&bnsum[nt1 * 16 + r], s1);
    atomicAdd(&bnsq [nt1 * 16 + r], q1);
  }
}

// ---------------------------------------------------------------------------
// Kernel 2: finalize BN statistics -> per-channel scale/shift.
// ---------------------------------------------------------------------------
__global__ void kpconv_stats(const float* __restrict__ stats,
                             const float* __restrict__ gamma,
                             const float* __restrict__ beta,
                             float* __restrict__ ss) {
  int o = threadIdx.x;
  if (o < COUT) {
    const float invn = 1.0f / (float)NPTS;
    float mu  = stats[o] * invn;
    float var = stats[COUT + o] * invn - mu * mu;
    float sc  = gamma[o] * rsqrtf(var + BN_EPS);
    ss[o]        = sc;
    ss[COUT + o] = beta[o] - mu * sc;
  }
}

// ---------------------------------------------------------------------------
// Kernel 3: apply BN affine + LeakyReLU in place (float4 vectorized).
// ---------------------------------------------------------------------------
__global__ __launch_bounds__(256) void kpconv_apply(float* __restrict__ out,
                                                    const float* __restrict__ ss) {
  __shared__ float sc[COUT], sh[COUT];
  if (threadIdx.x < COUT) {
    sc[threadIdx.x] = ss[threadIdx.x];
    sh[threadIdx.x] = ss[COUT + threadIdx.x];
  }
  __syncthreads();
  size_t i = ((size_t)blockIdx.x * blockDim.x + threadIdx.x) * 4;
  if (i < (size_t)NPTS * COUT) {
    float4 v = *(float4*)(out + i);
    int o = (int)(i & (COUT - 1));   // 4 | COUT: no row crossing
    float x0 = v.x * sc[o + 0] + sh[o + 0];
    float x1 = v.y * sc[o + 1] + sh[o + 1];
    float x2 = v.z * sc[o + 2] + sh[o + 2];
    float x3 = v.w * sc[o + 3] + sh[o + 3];
    v.x = (x0 >= 0.f) ? x0 : NEG_SLOPE * x0;
    v.y = (x1 >= 0.f) ? x1 : NEG_SLOPE * x1;
    v.z = (x2 >= 0.f) ? x2 : NEG_SLOPE * x2;
    v.w = (x3 >= 0.f) ? x3 : NEG_SLOPE * x3;
    *(float4*)(out + i) = v;
  }
}

// ---------------------------------------------------------------------------
extern "C" void kernel_launch(void* const* d_in, const int* in_sizes, int n_in,
                              void* d_out, int out_size, void* d_ws, size_t ws_size,
                              hipStream_t stream) {
  (void)in_sizes; (void)n_in; (void)out_size; (void)ws_size;
  const float* feats = (const float*)d_in[0];
  const float* xyz   = (const float*)d_in[1];
  // d_in[2] = batch (all zeros) unused
  const int*   nbr   = (const int*)d_in[3];
  const float* kpts  = (const float*)d_in[4];
  const float* w     = (const float*)d_in[5];
  const float* gamma = (const float*)d_in[6];
  const float* beta  = (const float*)d_in[7];
  float* out = (float*)d_out;

  // workspace layout: [sum:128][sumsq:128][scale:128][shift:128][wbf: 1024*128 bf16]
  float* ws    = (float*)d_ws;
  float* bnsum = ws;
  float* bnsq  = ws + COUT;
  float* ss    = ws + 2 * COUT;
  __bf16* wbf  = (__bf16*)(ws + 4 * COUT);

  kpconv_prep<<<(KRP * COUT + 255) / 256, 256, 0, stream>>>(w, wbf, bnsum);
  kpconv_main<<<NPTS / 16, 128, 0, stream>>>(feats, xyz, nbr, kpts, wbf, out, bnsum, bnsq);
  kpconv_stats<<<1, COUT, 0, stream>>>(bnsum, gamma, beta, ss);
  kpconv_apply<<<(NPTS * COUT / 4 + 255) / 256, 256, 0, stream>>>(out, ss);
}
